// TokenSwapMamba_81561428951253
// MI455X (gfx1250) — matrix operand order repaired and checked by reading, not compiled, not run on hardware
//
#include <hip/hip_runtime.h>
#include <hip/hip_bf16.h>
#include <math.h>

#define DIMC     96
#define D_INNER  192
#define D_STATE  16
#define DT_RANK  6
#define NUM_SWAP 59
#define BB       2
#define NN       6400
#define T_TOK    (BB * NN)          // 12800 tokens
#define XD       (DT_RANK + 2 * D_STATE)   // 38
#define CHUNKS   32
#define CLEN     (NN / CHUNKS)      // 200
#define STREAMS  (BB * D_INNER)     // 384

typedef __attribute__((ext_vector_type(16))) _Float16 v16h;
typedef __attribute__((ext_vector_type(8)))  _Float16 v8h;
typedef __attribute__((ext_vector_type(8)))  float    v8f;
typedef __attribute__((ext_vector_type(4)))  unsigned int u32x4;
typedef __attribute__((ext_vector_type(8)))  int      i32x8;
typedef __attribute__((ext_vector_type(4)))  int      i32x4;

// ---------------------------------------------------------------------------
// TDM helper: DMA a (rows x cols, f16, row stride "stride0" elements) tile
// from global memory into LDS at byte offset lds_off. 1-D copies pass rows=1.
// Falls back to nothing (caller does cooperative copy) if the builtin is absent.
// ---------------------------------------------------------------------------
#if __has_builtin(__builtin_amdgcn_tensor_load_to_lds)
#define HAS_TDM 1
__device__ __forceinline__ void tdm_load_f16(unsigned int lds_off,
                                             const _Float16* gptr,
                                             unsigned int cols, unsigned int rows,
                                             unsigned long long stride0) {
  const unsigned long long ga = (unsigned long long)(uintptr_t)gptr;
  u32x4 g0;
  g0.x = 1u;                                          // count=1, user descriptor
  g0.y = lds_off;                                     // D#.lds_addr
  g0.z = (unsigned int)(ga & 0xFFFFFFFFull);          // global_addr[31:0]
  g0.w = (unsigned int)((ga >> 32) & 0x01FFFFFFull) | (2u << 30);  // addr hi | type=2
  i32x8 g1;
  g1[0] = (int)(1u << 16);                            // data_size=1 (2 bytes)
  g1[1] = (int)((cols & 0xFFFFu) << 16);              // tensor_dim0[15:0] @ bits 63:48
  g1[2] = (int)(((cols >> 16) & 0xFFFFu) | (0x7FFFu << 16)); // dim0 hi | tensor_dim1 lo
  g1[3] = (int)(0x0u | ((cols & 0xFFFFu) << 16));     // tensor_dim1 hi=0 | tile_dim0=cols
  g1[4] = (int)(rows & 0xFFFFu);                      // tile_dim1 = rows, tile_dim2 = 0
  g1[5] = (int)(unsigned int)(stride0 & 0xFFFFFFFFull);   // tensor_dim0_stride lo
  g1[6] = (int)(unsigned int)((stride0 >> 32) & 0xFFFFull); // stride hi | dim1_stride lo
  g1[7] = 0;
  i32x4 z4 = {};
#if defined(__clang_major__) && (__clang_major__ >= 23)
  i32x8 z8 = {};
  __builtin_amdgcn_tensor_load_to_lds(g0, g1, z4, z4, z8, 0);
#else
  __builtin_amdgcn_tensor_load_to_lds(g0, g1, z4, z4, 0);
#endif
}
#else
#define HAS_TDM 0
#endif

// ---------------------------------------------------------------------------
// Kernel 1: LayerNorm + token swap + channel shuffle -> f16 activations
// ---------------------------------------------------------------------------
__global__ void ln_swap_shuffle(const float* __restrict__ x1, const float* __restrict__ x2,
                                const float* __restrict__ g1, const float* __restrict__ b1,
                                const float* __restrict__ g2, const float* __restrict__ b2,
                                _Float16* __restrict__ xs1, _Float16* __restrict__ xs2) {
  __shared__ float sh[8][2][DIMC];
  const int lane = threadIdx.x & 31;
  const int w    = threadIdx.x >> 5;
  const int t    = blockIdx.x * 8 + w;
  if (t >= T_TOK) return;
  const float* p1 = x1 + (size_t)t * DIMC;
  const float* p2 = x2 + (size_t)t * DIMC;
  float a[3], c[3];
  #pragma unroll
  for (int i = 0; i < 3; ++i) { a[i] = p1[lane * 3 + i]; c[i] = p2[lane * 3 + i]; }
  float s1 = a[0] + a[1] + a[2], s2 = c[0] + c[1] + c[2];
  #pragma unroll
  for (int m = 16; m >= 1; m >>= 1) { s1 += __shfl_xor(s1, m, 32); s2 += __shfl_xor(s2, m, 32); }
  const float m1 = s1 * (1.0f / DIMC), m2 = s2 * (1.0f / DIMC);
  float q1 = 0.f, q2 = 0.f;
  #pragma unroll
  for (int i = 0; i < 3; ++i) {
    q1 += (a[i] - m1) * (a[i] - m1);
    q2 += (c[i] - m2) * (c[i] - m2);
  }
  #pragma unroll
  for (int m = 16; m >= 1; m >>= 1) { q1 += __shfl_xor(q1, m, 32); q2 += __shfl_xor(q2, m, 32); }
  const float r1 = rsqrtf(q1 * (1.0f / DIMC) + 1e-5f);
  const float r2 = rsqrtf(q2 * (1.0f / DIMC) + 1e-5f);
  #pragma unroll
  for (int i = 0; i < 3; ++i) {
    const int ch = lane * 3 + i;
    sh[w][0][ch] = (a[i] - m1) * r1 * g1[ch] + b1[ch];
    sh[w][1][ch] = (c[i] - m2) * r2 * g2[ch] + b2[ch];
  }
  __syncthreads();
  #pragma unroll
  for (int i = 0; i < 3; ++i) {
    const int k   = lane * 3 + i;
    const int src = (k & 1) * 48 + (k >> 1);           // channel shuffle (groups=2)
    const float x1n = sh[w][0][src];
    const float x2n = sh[w][1][src];
    const float v1  = (src < NUM_SWAP) ? x2n : x1n;    // x1s pre-shuffle
    const float v2  = (src < NUM_SWAP) ? v1  : x2n;    // x2s pre-shuffle
    xs1[(size_t)t * DIMC + k] = (_Float16)v1;
    xs2[(size_t)t * DIMC + k] = (_Float16)v2;
  }
}

// ---------------------------------------------------------------------------
// Kernel 2: pack weight (Nrows x K, f32) into WMMA B-operand lane layout
// bp[(nt*KT+kt)*32 + lane][h] = W[nt*16 + lane%16][kt*32 + (lane/16)*16 + h]
// ---------------------------------------------------------------------------
__global__ void pack_b(const float* __restrict__ W, _Float16* __restrict__ bp,
                       int NT, int KT, int ldw) {
  const int idx = blockIdx.x * blockDim.x + threadIdx.x;
  if (idx >= NT * KT * 32) return;
  const int lane = idx & 31;
  const int kt   = (idx >> 5) % KT;
  const int nt   = (idx >> 5) / KT;
  const int j    = nt * 16 + (lane & 15);
  const int kb   = kt * 32 + (lane >> 4) * 16;
  _Float16* dst  = bp + ((size_t)(nt * KT + kt) * 32 + lane) * 16;
  #pragma unroll
  for (int h = 0; h < 16; ++h) dst[h] = (_Float16)W[(size_t)j * ldw + kb + h];
}

// ---------------------------------------------------------------------------
// Kernel 3: WMMA GEMM, LDS-staged (TDM -> LDS -> ds_load -> v_wmma)
//   block = 4 waves, computes 64 rows x (NB*16) cols
//   wave w : rows mblock*64 + w*16 .. +15, all NB n-tiles (A frag reused)
// ---------------------------------------------------------------------------
template <int KT, int NB, int LDA>
__global__ void wmma_gemm_lds(const _Float16* __restrict__ A,
                              const _Float16* __restrict__ Bp,
                              float* __restrict__ C, int ldc,
                              const float* __restrict__ residual) {
  __shared__ __align__(16) _Float16 shA[64 * LDA];
  __shared__ __align__(16) _Float16 shB[NB * KT * 512];

  const int lane   = threadIdx.x & 31;
  const int w      = threadIdx.x >> 5;
  const int mblock = blockIdx.x;
  const int ngrp   = blockIdx.y;

  const _Float16* gA = A  + (size_t)mblock * 64 * LDA;
  const _Float16* gB = Bp + (size_t)ngrp * NB * KT * 512;

#if HAS_TDM
  if (threadIdx.x < 32) {   // wave 0 drives the Tensor Data Mover
    tdm_load_f16((unsigned int)(uintptr_t)shA, gA, LDA, 64, LDA);
    tdm_load_f16((unsigned int)(uintptr_t)shB, gB, NB * KT * 512, 1, NB * KT * 512);
    __builtin_amdgcn_s_wait_tensorcnt(0);
  }
#else
  {  // cooperative fallback: 16B per thread per step
    const uint4* sa = (const uint4*)gA;  uint4* da = (uint4*)shA;
    const int na = 64 * LDA * 2 / 16;
    for (int i = threadIdx.x; i < na; i += 128) da[i] = sa[i];
    const uint4* sb = (const uint4*)gB;  uint4* db = (uint4*)shB;
    const int nb = NB * KT * 512 * 2 / 16;
    for (int i = threadIdx.x; i < nb; i += 128) db[i] = sb[i];
  }
#endif
  __syncthreads();

  const int lrow  = w * 16 + (lane & 15);
  const int kbase = (lane >> 4) * 16;

  v8f acc[NB];
  #pragma unroll
  for (int nb = 0; nb < NB; ++nb) acc[nb] = (v8f){};

  #pragma unroll
  for (int kt = 0; kt < KT; ++kt) {
    // A fragment from LDS: halves 0..15 = K kbase..kbase+15 (contiguous)
    const _Float16* ap = shA + lrow * LDA + kt * 32 + kbase;
    const v8h lo = *(const v8h*)(ap);
    const v8h hi = *(const v8h*)(ap + 8);
    v16h a;
    #pragma unroll
    for (int i = 0; i < 8; ++i) { a[i] = lo[i]; a[8 + i] = hi[i]; }
    #pragma unroll
    for (int nb = 0; nb < NB; ++nb) {
      const v16h b = *(const v16h*)(shB + ((nb * KT + kt) * 32 + lane) * 16);
      acc[nb] = __builtin_amdgcn_wmma_f32_16x16x32_f16(false, a, false, b, (short)0,
                                                       acc[nb], false, false);
    }
  }

  const int m0 = mblock * 64 + w * 16 + 8 * (lane >> 4);
  #pragma unroll
  for (int nb = 0; nb < NB; ++nb) {
    const int n = (ngrp * NB + nb) * 16 + (lane & 15);
    #pragma unroll
    for (int r = 0; r < 8; ++r) {
      float v = acc[nb][r];
      if (residual) v += residual[(size_t)(m0 + r) * ldc + n];
      C[(size_t)(m0 + r) * ldc + n] = v;
    }
  }
}

// ---------------------------------------------------------------------------
// Kernel 4: causal depthwise conv(4) + bias + SiLU
// ---------------------------------------------------------------------------
__global__ void conv_silu(const float* __restrict__ xz, const float* __restrict__ cw,
                          const float* __restrict__ cb, float* __restrict__ xca) {
  const int idx = blockIdx.x * blockDim.x + threadIdx.x;
  if (idx >= T_TOK * D_INNER) return;
  const int t = idx / D_INNER;
  const int d = idx - t * D_INNER;
  const int n = t % NN;
  float acc = cb[d];
  #pragma unroll
  for (int k = 0; k < 4; ++k) {
    const int nk = n - 3 + k;
    if (nk >= 0) acc += cw[d * 4 + k] * xz[(size_t)(t - 3 + k) * 384 + d];
  }
  xca[idx] = acc / (1.f + __expf(-acc));   // silu
}

// ---------------------------------------------------------------------------
// Kernel 5: x_dbl = xc_act @ xproj_w^T  (192 -> 38)
// ---------------------------------------------------------------------------
__global__ void xproj(const float* __restrict__ xca, const float* __restrict__ W,
                      float* __restrict__ xdbl) {
  const int idx = blockIdx.x * blockDim.x + threadIdx.x;
  if (idx >= T_TOK * XD) return;
  const int t = idx / XD;
  const int j = idx - t * XD;
  const float* xr = xca + (size_t)t * D_INNER;
  const float* wr = W + (size_t)j * D_INNER;
  float acc = 0.f;
  #pragma unroll 4
  for (int k = 0; k < D_INNER; ++k) acc += xr[k] * wr[k];
  xdbl[idx] = acc;
}

// ---------------------------------------------------------------------------
// Kernel 6: dt = softplus(dt_r @ dt_w^T + dt_b)
// ---------------------------------------------------------------------------
__global__ void dt_softplus(const float* __restrict__ xdbl, const float* __restrict__ dtw,
                            const float* __restrict__ dtb, float* __restrict__ dt) {
  const int idx = blockIdx.x * blockDim.x + threadIdx.x;
  if (idx >= T_TOK * D_INNER) return;
  const int t = idx / D_INNER;
  const int d = idx - t * D_INNER;
  float acc = dtb[d];
  #pragma unroll
  for (int r = 0; r < DT_RANK; ++r) acc += xdbl[(size_t)t * XD + r] * dtw[d * DT_RANK + r];
  dt[idx] = (acc > 20.f) ? acc : log1pf(__expf(acc));
}

// ---------------------------------------------------------------------------
// Kernel 7: scan phase 1 — per-chunk summaries (P = prod a, H = h from zero)
// ---------------------------------------------------------------------------
__global__ void scan_phase1(const float* __restrict__ dt, const float* __restrict__ xca,
                            const float* __restrict__ xdbl, const float* __restrict__ A_log,
                            float* __restrict__ Pout, float* __restrict__ Hout) {
  const int idx = blockIdx.x * blockDim.x + threadIdx.x;
  if (idx >= STREAMS * CHUNKS) return;
  const int chunk = idx & (CHUNKS - 1);
  const int st    = idx >> 5;            // CHUNKS == 32
  const int b = st / D_INNER, d = st - b * D_INNER;
  float A[D_STATE], h[D_STATE], P[D_STATE];
  #pragma unroll
  for (int s = 0; s < D_STATE; ++s) {
    A[s] = -__expf(A_log[d * D_STATE + s]);
    h[s] = 0.f; P[s] = 1.f;
  }
  const int n0 = chunk * CLEN;
  for (int i = 0; i < CLEN; ++i) {
    const size_t t = (size_t)b * NN + n0 + i;
    if (i + 8 < CLEN) {
      __builtin_prefetch(&dt[(t + 8) * D_INNER + d], 0, 0);
      __builtin_prefetch(&xca[(t + 8) * D_INNER + d], 0, 0);
    }
    const float dv = dt[t * D_INNER + d];
    const float u  = dv * xca[t * D_INNER + d];
    const float* Bv = xdbl + t * XD + DT_RANK;
    #pragma unroll
    for (int s = 0; s < D_STATE; ++s) {
      const float a = __expf(dv * A[s]);
      h[s] = a * h[s] + u * Bv[s];
      P[s] *= a;
    }
  }
  float* pp = Pout + (size_t)idx * D_STATE;
  float* hp = Hout + (size_t)idx * D_STATE;
  #pragma unroll
  for (int s = 0; s < D_STATE; ++s) { pp[s] = P[s]; hp[s] = h[s]; }
}

// ---------------------------------------------------------------------------
// Kernel 8: scan middle — thread chunk states through per stream
// ---------------------------------------------------------------------------
__global__ void scan_mid(const float* __restrict__ P, const float* __restrict__ H,
                         float* __restrict__ Hinit) {
  const int st = blockIdx.x * blockDim.x + threadIdx.x;
  if (st >= STREAMS) return;
  float h[D_STATE];
  #pragma unroll
  for (int s = 0; s < D_STATE; ++s) h[s] = 0.f;
  for (int c = 0; c < CHUNKS; ++c) {
    const size_t o = ((size_t)st * CHUNKS + c) * D_STATE;
    #pragma unroll
    for (int s = 0; s < D_STATE; ++s) Hinit[o + s] = h[s];
    #pragma unroll
    for (int s = 0; s < D_STATE; ++s) h[s] = P[o + s] * h[s] + H[o + s];
  }
}

// ---------------------------------------------------------------------------
// Kernel 9: scan phase 2 — replay chunk with true init, emit gated y (f16)
// ---------------------------------------------------------------------------
__global__ void scan_phase2(const float* __restrict__ dt, const float* __restrict__ xca,
                            const float* __restrict__ xdbl, const float* __restrict__ A_log,
                            const float* __restrict__ Dp, const float* __restrict__ xz,
                            const float* __restrict__ Hinit, _Float16* __restrict__ yh) {
  const int idx = blockIdx.x * blockDim.x + threadIdx.x;
  if (idx >= STREAMS * CHUNKS) return;
  const int chunk = idx & (CHUNKS - 1);
  const int st    = idx >> 5;
  const int b = st / D_INNER, d = st - b * D_INNER;
  float A[D_STATE], h[D_STATE];
  const float* hp = Hinit + (size_t)idx * D_STATE;
  #pragma unroll
  for (int s = 0; s < D_STATE; ++s) {
    A[s] = -__expf(A_log[d * D_STATE + s]);
    h[s] = hp[s];
  }
  const float Dd = Dp[d];
  const int n0 = chunk * CLEN;
  for (int i = 0; i < CLEN; ++i) {
    const size_t t = (size_t)b * NN + n0 + i;
    if (i + 8 < CLEN) {
      __builtin_prefetch(&dt[(t + 8) * D_INNER + d], 0, 0);
      __builtin_prefetch(&xz[(t + 8) * 384 + 192 + d], 0, 0);
    }
    const float dv = dt[t * D_INNER + d];
    const float xv = xca[t * D_INNER + d];
    const float u  = dv * xv;
    const float* Bv = xdbl + t * XD + DT_RANK;
    const float* Cv = xdbl + t * XD + DT_RANK + D_STATE;
    float acc = 0.f;
    #pragma unroll
    for (int s = 0; s < D_STATE; ++s) {
      const float a = __expf(dv * A[s]);
      h[s] = a * h[s] + u * Bv[s];
      acc += h[s] * Cv[s];
    }
    const float y = acc + xv * Dd;
    const float z = xz[t * 384 + 192 + d];
    yh[t * D_INNER + d] = (_Float16)(y * z / (1.f + __expf(-z)));
  }
}

// ---------------------------------------------------------------------------
// host-side launch
// ---------------------------------------------------------------------------
static inline size_t al256(size_t x) { return (x + 255) & ~(size_t)255; }

extern "C" void kernel_launch(void* const* d_in, const int* in_sizes, int n_in,
                              void* d_out, int out_size, void* d_ws, size_t ws_size,
                              hipStream_t stream) {
  (void)in_sizes; (void)n_in; (void)out_size; (void)ws_size;
  const float* x1   = (const float*)d_in[0];
  const float* x2   = (const float*)d_in[1];
  const float* ln1g = (const float*)d_in[2];
  const float* ln1b = (const float*)d_in[3];
  const float* ln2g = (const float*)d_in[4];
  const float* ln2b = (const float*)d_in[5];

  // ---- workspace carve ----
  char* ws = (char*)d_ws;
  size_t off = 0;
  auto carve = [&](size_t bytes) { char* p = ws + off; off = al256(off + bytes); return p; };

  _Float16* xs1h = (_Float16*)carve((size_t)T_TOK * DIMC * 2);
  _Float16* xs2h = (_Float16*)carve((size_t)T_TOK * DIMC * 2);
  _Float16* inpk[2]  = { (_Float16*)carve(24 * 3 * 32 * 16 * 2),
                         (_Float16*)carve(24 * 3 * 32 * 16 * 2) };
  _Float16* outpk[2] = { (_Float16*)carve(6 * 6 * 32 * 16 * 2),
                         (_Float16*)carve(6 * 6 * 32 * 16 * 2) };
  // per-expert scratch (reused: experts run sequentially on the stream)
  float*    xz    = (float*)carve((size_t)T_TOK * 384 * 4);
  float*    xca   = (float*)carve((size_t)T_TOK * D_INNER * 4);
  float*    xdbl  = (float*)carve((size_t)T_TOK * XD * 4);
  float*    dtbuf = (float*)carve((size_t)T_TOK * D_INNER * 4);
  _Float16* yh    = (_Float16*)carve((size_t)T_TOK * D_INNER * 2);
  float*    Pbuf  = (float*)carve((size_t)STREAMS * CHUNKS * D_STATE * 4);
  float*    Hbuf  = (float*)carve((size_t)STREAMS * CHUNKS * D_STATE * 4);
  float*    Hinit = (float*)carve((size_t)STREAMS * CHUNKS * D_STATE * 4);

  // ---- stage A: LN + swap + shuffle ----
  ln_swap_shuffle<<<T_TOK / 8, 256, 0, stream>>>(x1, x2, ln1g, ln1b, ln2g, ln2b, xs1h, xs2h);

  // ---- pack weights (both experts) ----
  for (int e = 0; e < 2; ++e) {
    const float* in_w  = (const float*)d_in[6 + 9 * e];
    const float* out_w = (const float*)d_in[14 + 9 * e];
    pack_b<<<(24 * 3 * 32 + 255) / 256, 256, 0, stream>>>(in_w, inpk[e], 24, 3, DIMC);
    pack_b<<<(6 * 6 * 32 + 255) / 256, 256, 0, stream>>>(out_w, outpk[e], 6, 6, D_INNER);
  }

  // ---- per-expert pipeline ----
  for (int e = 0; e < 2; ++e) {
    const float* conv_w  = (const float*)d_in[7 + 9 * e];
    const float* conv_b  = (const float*)d_in[8 + 9 * e];
    const float* xproj_w = (const float*)d_in[9 + 9 * e];
    const float* dt_w    = (const float*)d_in[10 + 9 * e];
    const float* dt_b    = (const float*)d_in[11 + 9 * e];
    const float* A_log   = (const float*)d_in[12 + 9 * e];
    const float* Dvec    = (const float*)d_in[13 + 9 * e];
    const _Float16* xs   = e ? xs2h : xs1h;
    const float* resid   = e ? x2 : x1;
    float* outp          = (float*)d_out + (size_t)e * T_TOK * DIMC;

    // in-proj: (T x 96) x (96 x 384) -> xz
    // grid: 200 row-blocks (64 rows) x 6 col-groups (64 cols), 4 waves/block
    wmma_gemm_lds<3, 4, DIMC><<<dim3(T_TOK / 64, 24 / 4), 128, 0, stream>>>(
        xs, inpk[e], xz, /*ldc=*/384, nullptr);

    conv_silu<<<(T_TOK * D_INNER) / 256, 256, 0, stream>>>(xz, conv_w, conv_b, xca);
    xproj<<<(T_TOK * XD + 255) / 256, 256, 0, stream>>>(xca, xproj_w, xdbl);
    dt_softplus<<<(T_TOK * D_INNER) / 256, 256, 0, stream>>>(xdbl, dt_w, dt_b, dtbuf);

    scan_phase1<<<(STREAMS * CHUNKS) / 256, 256, 0, stream>>>(dtbuf, xca, xdbl, A_log,
                                                              Pbuf, Hbuf);
    scan_mid<<<(STREAMS + 127) / 128, 128, 0, stream>>>(Pbuf, Hbuf, Hinit);
    scan_phase2<<<(STREAMS * CHUNKS) / 256, 256, 0, stream>>>(dtbuf, xca, xdbl, A_log,
                                                              Dvec, xz, Hinit, yh);

    // out-proj + residual: (T x 192) x (192 x 96) -> d_out
    // grid: 200 row-blocks x 2 col-groups (48 cols each)
    wmma_gemm_lds<6, 3, D_INNER><<<dim3(T_TOK / 64, 96 / 48), 128, 0, stream>>>(
        yh, outpk[e], outp, /*ldc=*/DIMC, resid);
  }
}
